// JointGenerator_54984171323968
// MI455X (gfx1250) — matrix operand order repaired and checked by reading, not compiled, not run on hardware
//
#include <hip/hip_runtime.h>
#include <hip/hip_bf16.h>

// ---------------------------------------------------------------------------
// Problem constants
// ---------------------------------------------------------------------------
#define B_  128
#define T_  256
#define H_  512
#define NL_ 3
#define G4H 2048              // 4*H
#define BT  (B_ * T_)         // 32768
#define NWG_REC 32            // persistent workgroups for the recurrent kernel

typedef __attribute__((ext_vector_type(16))) __bf16        v16bf;
typedef __attribute__((ext_vector_type(8)))  float         v8f;
typedef __attribute__((ext_vector_type(4)))  unsigned int  u32x4;
typedef __attribute__((ext_vector_type(4)))  float         f32x4;

// Fragment: 16 bf16 per lane == 32 bytes == two 128-bit loads.
union ABfrag { v16bf v; u32x4 q[2]; unsigned short u[16]; };

// Per-cell element offsets of the six swizzled weight blocks inside one
// contiguous allocation (sizes are exact multiples of the 256B align step).
#define WSZ0 (1152u * 2048u)
#define WSZ1 (1536u * 2048u)
__device__ __forceinline__ unsigned woff_of(int cell) {
    const unsigned o1 = WSZ0;
    const unsigned o2 = o1 + WSZ1;
    const unsigned o3 = o2 + WSZ1;
    const unsigned o4 = o3 + WSZ0;
    const unsigned o5 = o4 + WSZ1;
    unsigned off = 0u;
    if (cell == 1) off = o1;
    else if (cell == 2) off = o2;
    else if (cell == 3) off = o3;
    else if (cell == 4) off = o4;
    else if (cell == 5) off = o5;
    return off;
}

// ---------------------------------------------------------------------------
// Helpers
// ---------------------------------------------------------------------------
__device__ __forceinline__ unsigned short f2bf(float f) {
    unsigned u = __float_as_uint(f);
    unsigned r = u + 0x7FFFu + ((u >> 16) & 1u);   // round-to-nearest-even
    return (unsigned short)(r >> 16);
}
__device__ __forceinline__ float sigmoidf(float x) {
    return 1.0f / (1.0f + __expf(-x));
}

// ---------------------------------------------------------------------------
// Sense-reversing global barrier across all resident workgroups.
// ---------------------------------------------------------------------------
__device__ __forceinline__ void gbar(unsigned* cnt, unsigned* sense,
                                     unsigned target, unsigned nwg) {
    __syncthreads();
    if (threadIdx.x == 0) {
        __threadfence();
        unsigned arrived = atomicAdd(cnt, 1u) + 1u;
        if (arrived == nwg) {
            atomicExch(cnt, 0u);
            __threadfence();
            atomicExch(sense, target);
        } else {
            while (*((volatile unsigned*)sense) < target) {
                __builtin_amdgcn_s_sleep(1);
            }
        }
    }
    __syncthreads();
}

// ---------------------------------------------------------------------------
// Prep kernels
// ---------------------------------------------------------------------------
__global__ void __launch_bounds__(256) k_zero32(unsigned* p, size_t n) {
    for (size_t i = (size_t)blockIdx.x * 256 + threadIdx.x; i < n;
         i += (size_t)gridDim.x * 256) p[i] = 0u;
}

// w: [N][K] row-major fp32 -> wsw: block-swizzled bf16 in WMMA B-fragment order.
// Block (kb = k/32, nb = n/16), flat block id = kb*(N/16)+nb;
// within block: lane = (n&15) | (((k>>4)&1)<<4), elem = k&15.
__global__ void __launch_bounds__(256) k_wt_swz(const float* __restrict__ w,
                                                unsigned short* __restrict__ wsw,
                                                int N, int K) {
    const size_t tot = (size_t)N * K;
    for (size_t i = (size_t)blockIdx.x * 256 + threadIdx.x; i < tot;
         i += (size_t)gridDim.x * 256) {
        const int n = (int)(i / K), k = (int)(i % K);
        const unsigned blk  = (unsigned)(k >> 5) * (unsigned)(N >> 4) + (unsigned)(n >> 4);
        const unsigned lane = (unsigned)((n & 15) | (((k >> 4) & 1) << 4));
        const unsigned elem = (unsigned)(k & 15);
        wsw[(((blk << 5) + lane) << 4) + elem] = f2bf(w[i]);
    }
}

// noise (B,T,nd) fp32 -> x (T,B,nd) bf16
__global__ void __launch_bounds__(256) k_noise_conv(const float* __restrict__ src,
                                                    unsigned short* __restrict__ dst,
                                                    int nd) {
    const size_t tot = (size_t)B_ * T_ * nd;
    for (size_t i = (size_t)blockIdx.x * 256 + threadIdx.x; i < tot;
         i += (size_t)gridDim.x * 256) {
        const size_t b = i / ((size_t)T_ * nd);
        const size_t r = i % ((size_t)T_ * nd);
        const size_t t = r / nd, j = r % nd;
        dst[(t * B_ + b) * nd + j] = f2bf(src[i]);
    }
}

__global__ void __launch_bounds__(256) k_f2bf(const float* __restrict__ src,
                                              unsigned short* __restrict__ dst, size_t n) {
    for (size_t i = (size_t)blockIdx.x * 256 + threadIdx.x; i < n;
         i += (size_t)gridDim.x * 256) dst[i] = f2bf(src[i]);
}

// ---------------------------------------------------------------------------
// Fragment loads. All offsets are 32-bit elements from a uniform base so the
// compiler emits global_load_b128 (SGPR base + VGPR offset) / ds_load_b128.
// ---------------------------------------------------------------------------
__device__ __forceinline__ void load_a_frag_lds(const unsigned short* ldsA,
                                                int mtile, int lane, ABfrag& a) {
    const u32x4* p = (const u32x4*)(ldsA + ((unsigned)((mtile << 5) + lane) << 4));
    a.q[0] = p[0];
    a.q[1] = p[1];
}
// elemOff = base element offset of the 32x16 block (multiple of 512)
__device__ __forceinline__ void load_b_frag_g(const unsigned short* __restrict__ wsw,
                                              unsigned elemOff, int lane, ABfrag& b) {
    const u32x4* p = (const u32x4*)(wsw + elemOff + ((unsigned)lane << 4));
    b.q[0] = p[0];
    b.q[1] = p[1];
}

// LDS offset (in u16 elements) of the 16-byte chunk (row, ch) of an A slab.
__device__ __forceinline__ unsigned a_lds_off(int row, int ch) {
    const int mt = row >> 4, m = row & 15;
    return ((unsigned)(((mt << 5) + m + ((ch >> 1) << 4)) << 4)) + (unsigned)((ch & 1) << 3);
}

// Stage one 128x32 bf16 A-slab into LDS in WMMA A-fragment order.
// Exactly 2 chunk moves per thread (branch-free): global_load_b128 -> ds_store_b128.
__device__ __forceinline__ void stage_a_slab(unsigned short* ldsA,
                                             const unsigned short* __restrict__ src,
                                             unsigned rowStride, unsigned kloc, int tid) {
    const int r0 = tid >> 2, ch = tid & 3;
    const int r1 = r0 + 64;
    const u32x4 v0 = *(const u32x4*)(src + (unsigned)r0 * rowStride + kloc + (unsigned)(ch << 3));
    const u32x4 v1 = *(const u32x4*)(src + (unsigned)r1 * rowStride + kloc + (unsigned)(ch << 3));
    *(u32x4*)(ldsA + a_lds_off(r0, ch)) = v0;
    *(u32x4*)(ldsA + a_lds_off(r1, ch)) = v1;
}

// Uniform segment select for a 32-wide K slab of the concat input
// [x | h_self | h_cpl] (all segment boundaries are multiples of 32).
__device__ __forceinline__ void pick_seg(int kt, int xdim,
                                         const unsigned short* segx,
                                         const unsigned short* segh,
                                         const unsigned short* segc,
                                         const unsigned short*& src,
                                         unsigned& strd, unsigned& kloc) {
    if (kt < xdim)           { src = segx; strd = (unsigned)xdim; kloc = (unsigned)kt; }
    else if (kt < xdim + H_) { src = segh; strd = H_; kloc = (unsigned)(kt - xdim); }
    else                     { src = segc; strd = H_; kloc = (unsigned)(kt - xdim - H_); }
}

#define WMMA4(A, S0, S1, S2, S3)                                                                  \
    acc0 = __builtin_amdgcn_wmma_f32_16x16x32_bf16(false, (A).v, false, (S0).v, (short)0, acc0, false, false); \
    acc1 = __builtin_amdgcn_wmma_f32_16x16x32_bf16(false, (A).v, false, (S1).v, (short)0, acc1, false, false); \
    acc2 = __builtin_amdgcn_wmma_f32_16x16x32_bf16(false, (A).v, false, (S2).v, (short)0, acc2, false, false); \
    acc3 = __builtin_amdgcn_wmma_f32_16x16x32_bf16(false, (A).v, false, (S3).v, (short)0, acc3, false, false)

#define BLOAD4(base, eoff, B0, B1, B2, B3)                \
    load_b_frag_g(base, (eoff) + 0u * 512u, lane, B0);    \
    load_b_frag_g(base, (eoff) + 1u * 512u, lane, B1);    \
    load_b_frag_g(base, (eoff) + 2u * 512u, lane, B2);    \
    load_b_frag_g(base, (eoff) + 3u * 512u, lane, B3)

// ---------------------------------------------------------------------------
// Persistent recurrent kernel: all T steps, 6 cells/step.
// Per cell: gates[128][2048] = [x | h_self | h_cpl](128xK) @ W^T(Kx2048)  (WMMA)
// Software-pipelined: double-buffered LDS A slabs + register B fragments.
// ---------------------------------------------------------------------------
__global__ void __launch_bounds__(256, 1) k_recurrent(
    const unsigned short* __restrict__ xc,    // [T][B][128] bf16
    const unsigned short* __restrict__ xd,    // [T][B][128] bf16
    const unsigned short* __restrict__ wbase, // all 6 swizzled weights, contiguous
    unsigned short* __restrict__ hc,          // [2][3][B][H] bf16
    unsigned short* __restrict__ hd,          // [2][3][B][H] bf16
    float* __restrict__ cc,                   // [3][B][H]
    float* __restrict__ cd,                   // [3][B][H]
    float* __restrict__ gates,                // [B][2048]
    float* __restrict__ cseq,                 // [B][T][H]
    float* __restrict__ dseq,                 // [B][T][H]
    unsigned* bar_cnt, unsigned* bar_sense) {

    __shared__ unsigned short ldsA0[128 * 32];   // swizzled A fragments (8KB)
    __shared__ unsigned short ldsA1[128 * 32];   // double buffer (8KB)

    const int wg = blockIdx.x;
    const int tid = threadIdx.x;
    const int wave = tid >> 5;
    const int lane = tid & 31;
    const int n0 = wg * 64;                      // this WG's 64-col slice of 2048

    const int Ks[3] = {128 + 2 * H_, H_ + 2 * H_, H_ + 2 * H_};   // 1152,1536,1536

    unsigned phase = 0;

    for (int t = 0; t < T_; ++t) {
        const int prev = t & 1;
        const int cur  = prev ^ 1;

        for (int cell = 0; cell < 6; ++cell) {
            const int stack = cell / 3;          // 0 = continuous, 1 = discrete
            const int layer = cell % 3;
            const int Ktot  = Ks[layer];         // multiple of 64
            const unsigned woff = woff_of(cell);

            // segment sources for A = [x | h_self | h_cpl]
            const unsigned short* segx;
            int xdim;
            if (layer == 0) {
                segx = (stack ? xd : xc) + (size_t)t * B_ * 128; xdim = 128;
            } else {
                segx = (stack ? hd : hc) + ((size_t)cur * 3 + (layer - 1)) * B_ * H_;
                xdim = H_;
            }
            const unsigned short* segh =
                (stack ? hd : hc) + ((size_t)prev * 3 + layer) * B_ * H_;
            // c-stack couples to PREV-step d_h; d-stack couples to CURRENT-step c_h
            const unsigned short* segc =
                stack ? (hc + ((size_t)cur  * 3 + layer) * B_ * H_)
                      : (hd + ((size_t)prev * 3 + layer) * B_ * H_);

            // B-block element offset for slab kt: woff + ((kt/32)*128 + n0/16) * 512
            const unsigned eoff0 = woff + (unsigned)(n0 >> 4) * 512u;

            // ---------------- GEMM phase (pipelined) ----------------
            v8f acc0 = {0,0,0,0,0,0,0,0}, acc1 = {0,0,0,0,0,0,0,0};
            v8f acc2 = {0,0,0,0,0,0,0,0}, acc3 = {0,0,0,0,0,0,0,0};
            ABfrag a0, a1;
            ABfrag p0b0, p0b1, p0b2, p0b3;       // B set 0
            ABfrag p1b0, p1b1, p1b2, p1b3;       // B set 1

            // prologue: slab 0 into lds0 / set0
            {
                const unsigned short* src; unsigned strd, kloc;
                pick_seg(0, xdim, segx, segh, segc, src, strd, kloc);
                stage_a_slab(ldsA0, src, strd, kloc, tid);
                BLOAD4(wbase, eoff0, p0b0, p0b1, p0b2, p0b3);
            }
            __syncthreads();

            for (int kt = 0; kt < Ktot; kt += 64) {
                // ---- half 1: consume lds0/set0, preload kt+32 into lds1/set1
                {
                    const unsigned short* src; unsigned strd, kloc;
                    pick_seg(kt + 32, xdim, segx, segh, segc, src, strd, kloc);
                    stage_a_slab(ldsA1, src, strd, kloc, tid);
                    BLOAD4(wbase, eoff0 + (unsigned)((kt + 32) >> 5) * (128u * 512u),
                           p1b0, p1b1, p1b2, p1b3);
                }
                load_a_frag_lds(ldsA0, wave, lane, a0);
                WMMA4(a0, p0b0, p0b1, p0b2, p0b3);
                __syncthreads();

                // ---- half 2: consume lds1/set1, preload kt+64 into lds0/set0
                if (kt + 64 < Ktot) {
                    const unsigned short* src; unsigned strd, kloc;
                    pick_seg(kt + 64, xdim, segx, segh, segc, src, strd, kloc);
                    stage_a_slab(ldsA0, src, strd, kloc, tid);
                    BLOAD4(wbase, eoff0 + (unsigned)((kt + 64) >> 5) * (128u * 512u),
                           p0b0, p0b1, p0b2, p0b3);
                }
                load_a_frag_lds(ldsA1, wave, lane, a1);
                WMMA4(a1, p1b0, p1b1, p1b2, p1b3);
                __syncthreads();
            }

            // store accumulators to gates scratch
            {
                const unsigned nbase = (unsigned)(lane & 15);
                const unsigned mbase = (unsigned)(wave * 16 + ((lane >> 4) << 3));
#pragma unroll
                for (int r = 0; r < 8; ++r) {
                    const unsigned row = (mbase + (unsigned)r) * G4H + (unsigned)n0;
                    gates[row + 0  + nbase] = acc0[r];
                    gates[row + 16 + nbase] = acc1[r];
                    gates[row + 32 + nbase] = acc2[r];
                    gates[row + 48 + nbase] = acc3[r];
                }
            }
            ++phase; gbar(bar_cnt, bar_sense, phase, NWG_REC);

            // ---------------- elementwise LSTM update ----------------
            float* cst = (stack ? cd : cc) + (size_t)layer * B_ * H_;
            unsigned short* hdst =
                (stack ? hd : hc) + ((size_t)cur * 3 + layer) * B_ * H_;
            float* seq = stack ? dseq : cseq;

            for (int e = wg * 256 + tid; e < B_ * H_; e += NWG_REC * 256) {
                const int bb = e >> 9;      // /512
                const int j  = e & 511;
                const unsigned g = (unsigned)bb * G4H + (unsigned)j;
                const float gi = gates[g];
                const float gf = gates[g + 512];
                const float go = gates[g + 1024];
                const float gc = gates[g + 1536];
                const float cn = sigmoidf(gf) * cst[e] + sigmoidf(gi) * tanhf(gc);
                const float hn = sigmoidf(go) * tanhf(cn);
                cst[e]  = cn;
                hdst[e] = f2bf(hn);
                if (layer == NL_ - 1)
                    seq[((size_t)bb * T_ + t) * H_ + j] = hn;
            }
            ++phase; gbar(bar_cnt, bar_sense, phase, NWG_REC);
        }
    }
}

// ---------------------------------------------------------------------------
// Generic WMMA GEMM: C[M][N] = A_bf16[M][K] @ Wswz_bf16 + bias[N]
// A row-major; W pre-swizzled (k_wt_swz). grid = (M/128, N/64), block = 256.
// K must be a multiple of 64 (here always 512). Same pipelined structure.
// ---------------------------------------------------------------------------
__global__ void __launch_bounds__(256, 1) k_gemm(const unsigned short* __restrict__ A,
                                                 const unsigned short* __restrict__ Wsw,
                                                 const float* __restrict__ bias,
                                                 float* __restrict__ C,
                                                 int M, int K, int N) {
    __shared__ unsigned short ldsA0[128 * 32];
    __shared__ unsigned short ldsA1[128 * 32];
    const int tid  = threadIdx.x;
    const int wave = tid >> 5;
    const int lane = tid & 31;
    const int m0 = blockIdx.x * 128;
    const int n0 = blockIdx.y * 64;
    const unsigned nblkRow = (unsigned)(N >> 4) * 512u;   // elements per K-slab of blocks

    v8f acc0 = {0,0,0,0,0,0,0,0}, acc1 = {0,0,0,0,0,0,0,0};
    v8f acc2 = {0,0,0,0,0,0,0,0}, acc3 = {0,0,0,0,0,0,0,0};
    ABfrag a0, a1;
    ABfrag p0b0, p0b1, p0b2, p0b3;
    ABfrag p1b0, p1b1, p1b2, p1b3;

    const unsigned short* Arow = A + (size_t)m0 * K;      // rows m0..m0+127
    const unsigned eoff0 = (unsigned)(n0 >> 4) * 512u;

    stage_a_slab(ldsA0, Arow, (unsigned)K, 0u, tid);
    BLOAD4(Wsw, eoff0, p0b0, p0b1, p0b2, p0b3);
    __syncthreads();

    for (int kt = 0; kt < K; kt += 64) {
        // half 1
        stage_a_slab(ldsA1, Arow, (unsigned)K, (unsigned)(kt + 32), tid);
        BLOAD4(Wsw, eoff0 + (unsigned)((kt + 32) >> 5) * nblkRow, p1b0, p1b1, p1b2, p1b3);
        load_a_frag_lds(ldsA0, wave, lane, a0);
        WMMA4(a0, p0b0, p0b1, p0b2, p0b3);
        __syncthreads();

        // half 2
        if (kt + 64 < K) {
            stage_a_slab(ldsA0, Arow, (unsigned)K, (unsigned)(kt + 64), tid);
            BLOAD4(Wsw, eoff0 + (unsigned)((kt + 64) >> 5) * nblkRow, p0b0, p0b1, p0b2, p0b3);
        }
        load_a_frag_lds(ldsA1, wave, lane, a1);
        WMMA4(a1, p1b0, p1b1, p1b2, p1b3);
        __syncthreads();
    }

    const int nbase = lane & 15;
    const int mbase = m0 + wave * 16 + ((lane >> 4) << 3);
#pragma unroll
    for (int r = 0; r < 8; ++r) {
        const size_t row = (size_t)(mbase + r) * N + n0;
        C[row + 0  + nbase] = acc0[r] + bias[n0 + 0  + nbase];
        C[row + 16 + nbase] = acc1[r] + bias[n0 + 16 + nbase];
        C[row + 32 + nbase] = acc2[r] + bias[n0 + 32 + nbase];
        C[row + 48 + nbase] = acc3[r] + bias[n0 + 48 + nbase];
    }
}

// ---------------------------------------------------------------------------
// Attention (faithful to torch .view semantics):
//   logits[b,t,s] = sum_k Qp[b, 4k+(t>>6), t&63] * Kp[b, 4k+(s>>6), s&63]
// ---------------------------------------------------------------------------
__global__ void __launch_bounds__(256) k_logits(const float* __restrict__ Qp,
                                                const float* __restrict__ Kp,
                                                float* __restrict__ attn) {
    const size_t idx = (size_t)blockIdx.x * 256 + threadIdx.x;   // over B*T*T
    const int s  = idx & 255;
    const size_t bt = idx >> 8;
    const int tq = bt & 255;
    const int b  = (int)(bt >> 8);
    const float* Qb = Qp + (size_t)b * T_ * 64;
    const float* Kb = Kp + (size_t)b * T_ * 64;
    float sum = 0.f;
#pragma unroll 8
    for (int k = 0; k < 64; ++k) {
        sum += Qb[(unsigned)((4 * k + (tq >> 6)) * 64 + (tq & 63))] *
               Kb[(unsigned)((4 * k + (s  >> 6)) * 64 + (s  & 63))];
    }
    attn[idx] = sum;
}

__global__ void __launch_bounds__(256) k_softmax(float* __restrict__ attn) {
    __shared__ float red[256];
    const int tid = threadIdx.x;
    float* p = attn + (size_t)blockIdx.x * 256;
    const float v = p[tid];
    red[tid] = v; __syncthreads();
    for (int st = 128; st > 0; st >>= 1) {
        if (tid < st) red[tid] = fmaxf(red[tid], red[tid + st]);
        __syncthreads();
    }
    const float mx = red[0]; __syncthreads();
    const float e = __expf(v - mx);
    red[tid] = e; __syncthreads();
    for (int st = 128; st > 0; st >>= 1) {
        if (tid < st) red[tid] += red[tid + st];
        __syncthreads();
    }
    p[tid] = e / red[0];
}

// Fused: out-matmul (view semantics) + gamma residual + bf16 convert for FC input
//   fcin[b,t,h] = bf16( gamma * sum_tt Vp[b, hh>>1, 256*(hh&1)+tt] * attn[b,ss,tt]
//                       + seq[b,t,h] ),  hh = 2t + (h>>8), ss = h & 255
__global__ void __launch_bounds__(256) k_attnout_fcin(const float* __restrict__ Vp,
                                                      const float* __restrict__ attn,
                                                      const float* __restrict__ seq,
                                                      const float* __restrict__ gamma,
                                                      unsigned short* __restrict__ fcin) {
    const size_t idx = (size_t)blockIdx.x * 256 + threadIdx.x;   // over B*T*H
    const int h  = idx & 511;
    const size_t bt = idx >> 9;
    const int t  = bt & 255;
    const int b  = (int)(bt >> 8);
    const int hh = 2 * t + (h >> 8);
    const int ss = h & 255;
    const f32x4* Vr = (const f32x4*)(Vp + ((size_t)b * T_ + (hh >> 1)) * H_ + ((hh & 1) << 8));
    const f32x4* Ar = (const f32x4*)(attn + ((size_t)b * T_ + ss) * T_);
    float sum = 0.f;
#pragma unroll 4
    for (int tt = 0; tt < T_ / 4; ++tt) {
        const f32x4 vv = Vr[tt], aa = Ar[tt];
        sum += vv[0] * aa[0] + vv[1] * aa[1] + vv[2] * aa[2] + vv[3] * aa[3];
    }
    fcin[idx] = f2bf(gamma[0] * sum + seq[idx]);
}

// ---------------------------------------------------------------------------
// Host-side orchestration
// ---------------------------------------------------------------------------
extern "C" void kernel_launch(void* const* d_in, const int* in_sizes, int n_in,
                              void* d_out, int out_size, void* d_ws, size_t ws_size,
                              hipStream_t stream) {
    (void)in_sizes; (void)n_in; (void)out_size; (void)ws_size;

    const float* noise_c = (const float*)d_in[0];
    const float* noise_d = (const float*)d_in[1];
    const float* cW[3] = {(const float*)d_in[2], (const float*)d_in[3], (const float*)d_in[4]};
    const float* dW[3] = {(const float*)d_in[14], (const float*)d_in[15], (const float*)d_in[16]};

    // workspace carve-out
    size_t off = 0;
    auto take = [&](size_t bytes) -> void* {
        off = (off + 255) & ~(size_t)255;
        void* p = (char*)d_ws + off;
        off += bytes;
        return p;
    };
    const int Ks[3] = {128 + 2 * H_, H_ + 2 * H_, H_ + 2 * H_};

    // all six cell-weight regions in one contiguous allocation (constant offsets)
    unsigned short* wbase = (unsigned short*)take(
        (size_t)2 * (Ks[0] + Ks[1] + Ks[2]) * G4H * 2);
    unsigned short* wt_c[3]; unsigned short* wt_d[3];
    {
        size_t o = 0;
        for (int i = 0; i < 3; ++i) { wt_c[i] = wbase + o; o += (size_t)Ks[i] * G4H; }
        for (int i = 0; i < 3; ++i) { wt_d[i] = wbase + o; o += (size_t)Ks[i] * G4H; }
    }
    unsigned short* xc = (unsigned short*)take((size_t)T_ * B_ * 128 * 2);
    unsigned short* xd = (unsigned short*)take((size_t)T_ * B_ * 128 * 2);
    unsigned short* hc = (unsigned short*)take((size_t)2 * 3 * B_ * H_ * 2);
    unsigned short* hd = (unsigned short*)take((size_t)2 * 3 * B_ * H_ * 2);
    float* cc    = (float*)take((size_t)3 * B_ * H_ * 4);
    float* cd    = (float*)take((size_t)3 * B_ * H_ * 4);
    float* gates = (float*)take((size_t)B_ * G4H * 4);
    unsigned* bar = (unsigned*)take(256);
    float* cseq  = (float*)take((size_t)BT * H_ * 4);
    float* dseq  = (float*)take((size_t)BT * H_ * 4);
    // attention scratch (reused for both stacks, processed sequentially)
    unsigned short* seq_bf = (unsigned short*)take((size_t)BT * H_ * 2);
    unsigned short* qwt  = (unsigned short*)take((size_t)H_ * 64 * 2);
    unsigned short* kwt  = (unsigned short*)take((size_t)H_ * 64 * 2);
    unsigned short* vwt  = (unsigned short*)take((size_t)H_ * H_ * 2);
    unsigned short* fcwt = (unsigned short*)take((size_t)H_ * 256 * 2);
    float* Qp   = (float*)take((size_t)BT * 64 * 4);
    float* Kp   = (float*)take((size_t)BT * 64 * 4);
    float* Vp   = (float*)take((size_t)BT * H_ * 4);
    float* attn = (float*)take((size_t)B_ * T_ * T_ * 4);
    unsigned short* fcin = (unsigned short*)take((size_t)BT * H_ * 2);

    // ---- prep: weight swizzle-convert, noise relayout, state zero ----
    for (int i = 0; i < 3; ++i) {
        k_wt_swz<<<2048, 256, 0, stream>>>(cW[i], wt_c[i], G4H, Ks[i]);
        k_wt_swz<<<2048, 256, 0, stream>>>(dW[i], wt_d[i], G4H, Ks[i]);
    }
    k_noise_conv<<<2048, 256, 0, stream>>>(noise_c, xc, 128);
    k_noise_conv<<<2048, 256, 0, stream>>>(noise_d, xd, 128);
    k_zero32<<<512, 256, 0, stream>>>((unsigned*)hc, (size_t)2 * 3 * B_ * H_ / 2);
    k_zero32<<<512, 256, 0, stream>>>((unsigned*)hd, (size_t)2 * 3 * B_ * H_ / 2);
    k_zero32<<<512, 256, 0, stream>>>((unsigned*)cc, (size_t)3 * B_ * H_);
    k_zero32<<<512, 256, 0, stream>>>((unsigned*)cd, (size_t)3 * B_ * H_);
    k_zero32<<<1, 64, 0, stream>>>(bar, 64);

    // ---- persistent recurrence over all T steps ----
    k_recurrent<<<NWG_REC, 256, 0, stream>>>(
        xc, xd, wbase, hc, hd, cc, cd, gates, cseq, dseq, bar, bar + 1);

    // ---- attention + FC, per stack ----
    for (int s = 0; s < 2; ++s) {
        const float* seq = s ? dseq : cseq;
        const int base = s ? 17 : 5;
        const float* fc_w  = (const float*)d_in[base + 0];
        const float* fc_b  = (const float*)d_in[base + 1];
        const float* q_w   = (const float*)d_in[base + 2];
        const float* q_b   = (const float*)d_in[base + 3];
        const float* k_w   = (const float*)d_in[base + 4];
        const float* k_b   = (const float*)d_in[base + 5];
        const float* v_w   = (const float*)d_in[base + 6];
        const float* v_b   = (const float*)d_in[base + 7];
        const float* gamma = (const float*)d_in[base + 8];
        float* out = (float*)d_out + (size_t)s * BT * 256;

        k_f2bf<<<4096, 256, 0, stream>>>(seq, seq_bf, (size_t)BT * H_);
        k_wt_swz<<<256, 256, 0, stream>>>(q_w,  qwt,  64,  H_);
        k_wt_swz<<<256, 256, 0, stream>>>(k_w,  kwt,  64,  H_);
        k_wt_swz<<<1024, 256, 0, stream>>>(v_w,  vwt,  H_,  H_);
        k_wt_swz<<<512, 256, 0, stream>>>(fc_w, fcwt, 256, H_);

        k_gemm<<<dim3(BT / 128, 1), 256, 0, stream>>>(seq_bf, qwt, q_b, Qp, BT, H_, 64);
        k_gemm<<<dim3(BT / 128, 1), 256, 0, stream>>>(seq_bf, kwt, k_b, Kp, BT, H_, 64);
        k_gemm<<<dim3(BT / 128, 8), 256, 0, stream>>>(seq_bf, vwt, v_b, Vp, BT, H_, H_);

        k_logits<<<(B_ * T_ * T_) / 256, 256, 0, stream>>>(Qp, Kp, attn);
        k_softmax<<<B_ * T_, 256, 0, stream>>>(attn);
        k_attnout_fcin<<<(BT * H_) / 256, 256, 0, stream>>>(Vp, attn, seq, gamma, fcin);

        k_gemm<<<dim3(BT / 128, 4), 256, 0, stream>>>(fcin, fcwt, fc_b, out, BT, H_, 256);
    }
}
